// Qwen3_5MoeSparseMoeBlock_50242527429149
// MI455X (gfx1250) — matrix operand order repaired and checked by reading, not compiled, not run on hardware
//
#include <hip/hip_runtime.h>

// ---------------------------------------------------------------------------
// Qwen3.5 MoE sparse block for MI455X (gfx1250), bf16 WMMA path.
// ---------------------------------------------------------------------------

typedef unsigned short ushort_t;
typedef __attribute__((ext_vector_type(16))) __bf16    v16bf;
typedef __attribute__((ext_vector_type(8)))  float     v8f;
typedef __attribute__((ext_vector_type(8)))  unsigned  v8u;

// Problem sizes (fixed by the reference)
constexpr int Td  = 4096;   // B*S tokens
constexpr int Hd  = 2048;
constexpr int Ed  = 8;
constexpr int Id  = 768;
constexpr int SHd = 5632;
constexpr int TWOI = 2 * Id;        // 1536
constexpr int KEd  = Ed * Id;       // 6144 (expert down K)
constexpr int NEd  = Ed * Id;       // 6144 (expert gate/up N)

constexpr int PF_ELEMS = 512;       // prefetch distance (bf16 elems ~ 1KB)

// ---------------------------------------------------------------------------
// helpers
// ---------------------------------------------------------------------------
__device__ __forceinline__ ushort_t f2bf(float f) {
    unsigned u = __builtin_bit_cast(unsigned, f);
    u += 0x7FFFu + ((u >> 16) & 1u);          // round-to-nearest-even
    return (ushort_t)(u >> 16);
}

// A fragment (16x32 bf16): per lane two contiguous 8-elem chunks at p and p+8
__device__ __forceinline__ v16bf load_a_frag(const ushort_t* p) {
    uint4 lo = *reinterpret_cast<const uint4*>(p);
    uint4 hi = *reinterpret_cast<const uint4*>(p + 8);
    v8u u;
    u[0]=lo.x; u[1]=lo.y; u[2]=lo.z; u[3]=lo.w;
    u[4]=hi.x; u[5]=hi.y; u[6]=hi.z; u[7]=hi.w;
    return __builtin_bit_cast(v16bf, u);
}

// B fragment (32x16 bf16, weights stored transposed): 16 contiguous bf16
__device__ __forceinline__ v16bf load_b_frag(const ushort_t* p) {
    uint4 lo = *reinterpret_cast<const uint4*>(p);
    uint4 hi = *reinterpret_cast<const uint4*>(p + 8);
    v8u u;
    u[0]=lo.x; u[1]=lo.y; u[2]=lo.z; u[3]=lo.w;
    u[4]=hi.x; u[5]=hi.y; u[6]=hi.z; u[7]=hi.w;
    return __builtin_bit_cast(v16bf, u);
}

__device__ __forceinline__ v8f wmma_bf16(v16bf a, v16bf b, v8f c) {
    return __builtin_amdgcn_wmma_f32_16x16x32_bf16(
        /*neg_a=*/false, a, /*neg_b=*/false, b,
        /*c_mod=*/(short)0, c, /*reuse_a=*/false, /*reuse_b=*/false);
}

// ---------------------------------------------------------------------------
// f32 -> bf16 straight conversion (4 elems / thread)
// ---------------------------------------------------------------------------
__global__ void k_cvt4(const float4* __restrict__ in, ushort_t* __restrict__ out, int n4) {
    int i = blockIdx.x * blockDim.x + threadIdx.x;
    if (i >= n4) return;
    float4 v = in[i];
    unsigned p0 = (unsigned)f2bf(v.x) | ((unsigned)f2bf(v.y) << 16);
    unsigned p1 = (unsigned)f2bf(v.z) | ((unsigned)f2bf(v.w) << 16);
    reinterpret_cast<uint2*>(out)[i] = make_uint2(p0, p1);
}

// ---------------------------------------------------------------------------
// Tiled transpose + f32->bf16: in[b][r][c] f32  ->  out[b][c][r] bf16
// rows, cols are multiples of 32. block(32,8), grid(cols/32, rows/32, batch)
// ---------------------------------------------------------------------------
__global__ void k_transpose_cvt(const float* __restrict__ in, ushort_t* __restrict__ out,
                                int rows, int cols) {
    __shared__ ushort_t tile[32][33];
    const size_t base = (size_t)blockIdx.z * rows * cols;
    const int c0 = blockIdx.x * 32, r0 = blockIdx.y * 32;
    #pragma unroll
    for (int i = 0; i < 4; ++i) {
        int r = r0 + threadIdx.y + i * 8;
        int c = c0 + threadIdx.x;
        tile[threadIdx.y + i * 8][threadIdx.x] = f2bf(in[base + (size_t)r * cols + c]);
    }
    __syncthreads();
    #pragma unroll
    for (int i = 0; i < 4; ++i) {
        int n = c0 + threadIdx.y + i * 8;   // output row = input col
        int k = r0 + threadIdx.x;           // output col = input row
        out[base + (size_t)n * rows + k] = tile[threadIdx.x][threadIdx.y + i * 8];
    }
}

// ---------------------------------------------------------------------------
// Router: one wave per token. logits = x@router_w, softmax, top-2 renorm,
// dense routing weights [E][T]; also sg[t] = sigmoid(x @ shared_expert_gate_w)
// ---------------------------------------------------------------------------
__global__ void k_router(const float* __restrict__ x, const float* __restrict__ rw,
                         const float* __restrict__ segw,
                         float* __restrict__ routing,   // [E*T]
                         float* __restrict__ sg,        // [T]
                         float* __restrict__ logits) {  // [T*E]
    const int t    = (blockIdx.x * blockDim.x + threadIdx.x) >> 5;
    const int lane = threadIdx.x & 31;
    if (t >= Td) return;
    const float* xr = x + (size_t)t * Hd;
    float acc[Ed];
    #pragma unroll
    for (int e = 0; e < Ed; ++e) acc[e] = 0.0f;
    float accg = 0.0f;
    for (int h = lane; h < Hd; h += 32) {
        float xv = xr[h];
        #pragma unroll
        for (int e = 0; e < Ed; ++e) acc[e] += xv * rw[(size_t)h * Ed + e];
        accg += xv * segw[h];
    }
    #pragma unroll
    for (int off = 16; off > 0; off >>= 1) {
        #pragma unroll
        for (int e = 0; e < Ed; ++e) acc[e] += __shfl_xor(acc[e], off, 32);
        accg += __shfl_xor(accg, off, 32);
    }
    if (lane == 0) {
        float m = acc[0];
        #pragma unroll
        for (int e = 1; e < Ed; ++e) m = fmaxf(m, acc[e]);
        float p[Ed], sum = 0.0f;
        #pragma unroll
        for (int e = 0; e < Ed; ++e) { p[e] = __expf(acc[e] - m); sum += p[e]; }
        float inv = 1.0f / sum;
        #pragma unroll
        for (int e = 0; e < Ed; ++e) p[e] *= inv;
        // top-2
        int i1 = 0;
        #pragma unroll
        for (int e = 1; e < Ed; ++e) if (p[e] > p[i1]) i1 = e;
        int i2 = (i1 == 0) ? 1 : 0;
        #pragma unroll
        for (int e = 0; e < Ed; ++e) if (e != i1 && p[e] > p[i2]) i2 = e;
        float denom = 1.0f / (p[i1] + p[i2]);
        float w1 = p[i1] * denom, w2 = p[i2] * denom;
        #pragma unroll
        for (int e = 0; e < Ed; ++e) {
            float r = (e == i1) ? w1 : ((e == i2) ? w2 : 0.0f);
            routing[(size_t)e * Td + t] = r;
            logits[(size_t)t * Ed + e]  = acc[e];
        }
        sg[t] = 1.0f / (1.0f + __expf(-accg));
    }
}

// ---------------------------------------------------------------------------
// Fused gate+up GEMM with SiLU epilogue.
//   Cg = A @ Bg, Cu = A @ Bu  (A: [T,K] bf16 row-major; B transposed: col n at
//   base + (n/Nblk)*blk_stride + (n%Nblk)*K + k)
//   H[t,n] = bf16( silu(Cg) * Cu * scale ),
//   scale = scale_blk ? scale_ptr[(n/scale_blk)*T + t] : scale_ptr[t]
// Workgroup: 256 thr = 8 waves, tile 64(M) x 128(N); wave tile 32x32 (x2 B).
// ---------------------------------------------------------------------------
__global__ __launch_bounds__(256)
void k_gateup(const ushort_t* __restrict__ A,
              const ushort_t* __restrict__ Bg, const ushort_t* __restrict__ Bu,
              const float* __restrict__ scale, int scale_blk,
              ushort_t* __restrict__ Hout,
              int Nblk, long long blk_stride, int K, int ldh) {
    const int lane = threadIdx.x & 31;
    const int w    = threadIdx.x >> 5;
    const int hf   = lane >> 4;        // half of wave
    const int l16  = lane & 15;
    const int m0 = blockIdx.y * 64  + (w >> 2) * 32;
    const int n0 = blockIdx.x * 128 + (w & 3) * 32;

    const ushort_t* pa0 = A + (size_t)(m0 + l16) * K + 16 * hf;
    const ushort_t* pa1 = pa0 + (size_t)16 * K;

    const int ng0 = n0 + l16, ng1 = n0 + 16 + l16;
    const int e0 = ng0 / Nblk, i0 = ng0 % Nblk;
    const int e1 = ng1 / Nblk, i1 = ng1 % Nblk;
    const ushort_t* pg0 = Bg + (size_t)e0 * blk_stride + (size_t)i0 * K + 16 * hf;
    const ushort_t* pg1 = Bg + (size_t)e1 * blk_stride + (size_t)i1 * K + 16 * hf;
    const ushort_t* pu0 = Bu + (size_t)e0 * blk_stride + (size_t)i0 * K + 16 * hf;
    const ushort_t* pu1 = Bu + (size_t)e1 * blk_stride + (size_t)i1 * K + 16 * hf;

    v8f cg00 = {}, cg01 = {}, cg10 = {}, cg11 = {};
    v8f cu00 = {}, cu01 = {}, cu10 = {}, cu11 = {};

    for (int k0 = 0; k0 < K; k0 += 32) {
        // stream prefetch ~1KB ahead (global_prefetch_b8, speculative-safe)
        __builtin_prefetch(pa0 + k0 + PF_ELEMS, 0, 1);
        __builtin_prefetch(pa1 + k0 + PF_ELEMS, 0, 1);
        __builtin_prefetch(pg0 + k0 + PF_ELEMS, 0, 1);
        __builtin_prefetch(pg1 + k0 + PF_ELEMS, 0, 1);
        __builtin_prefetch(pu0 + k0 + PF_ELEMS, 0, 1);
        __builtin_prefetch(pu1 + k0 + PF_ELEMS, 0, 1);

        v16bf a0  = load_a_frag(pa0 + k0);
        v16bf a1  = load_a_frag(pa1 + k0);
        v16bf bg0 = load_b_frag(pg0 + k0);
        v16bf bg1 = load_b_frag(pg1 + k0);
        v16bf bu0 = load_b_frag(pu0 + k0);
        v16bf bu1 = load_b_frag(pu1 + k0);
        cg00 = wmma_bf16(a0, bg0, cg00);
        cg01 = wmma_bf16(a0, bg1, cg01);
        cg10 = wmma_bf16(a1, bg0, cg10);
        cg11 = wmma_bf16(a1, bg1, cg11);
        cu00 = wmma_bf16(a0, bu0, cu00);
        cu01 = wmma_bf16(a0, bu1, cu01);
        cu10 = wmma_bf16(a1, bu0, cu10);
        cu11 = wmma_bf16(a1, bu1, cu11);
    }

    // epilogue: silu(g) * u * scale -> bf16
    v8f CG[2][2] = {{cg00, cg01}, {cg10, cg11}};
    v8f CU[2][2] = {{cu00, cu01}, {cu10, cu11}};
    #pragma unroll
    for (int mi = 0; mi < 2; ++mi) {
        #pragma unroll
        for (int ni = 0; ni < 2; ++ni) {
            const int t0 = m0 + mi * 16 + 8 * hf;
            const int n  = n0 + ni * 16 + l16;
            #pragma unroll
            for (int j = 0; j < 8; ++j) {
                const int t = t0 + j;
                float g = CG[mi][ni][j];
                float u = CU[mi][ni][j];
                float s = scale_blk ? scale[(size_t)(n / scale_blk) * Td + t]
                                    : scale[t];
                float sig = 1.0f / (1.0f + __expf(-g));
                Hout[(size_t)t * ldh + n] = f2bf(g * sig * u * s);
            }
        }
    }
}

// ---------------------------------------------------------------------------
// Down GEMM: C[T,N] (+)= A[T,K] @ B[K,N].  B stored transposed with K split
// into blocks of Kblk (per-expert): elem(k,n) at
//   BT + (k/Kblk)*kblk_stride + n*ldb + (k%Kblk)
// Workgroup 64(M) x 256(N), wave tile 32x64 (8 WMMA per 6 fragment loads).
// beta: 1 -> accumulate into C.
// ---------------------------------------------------------------------------
__global__ __launch_bounds__(256)
void k_down(const ushort_t* __restrict__ A, int lda,
            const ushort_t* __restrict__ BT, int ldb,
            int Kblk, long long kblk_stride, int K,
            float* __restrict__ C, int ldc, int beta) {
    const int lane = threadIdx.x & 31;
    const int w    = threadIdx.x >> 5;
    const int hf   = lane >> 4;
    const int l16  = lane & 15;
    const int m0 = blockIdx.y * 64  + (w >> 2) * 32;
    const int n0 = blockIdx.x * 256 + (w & 3) * 64;

    const ushort_t* pa0 = A + (size_t)(m0 + l16) * lda + 16 * hf;
    const ushort_t* pa1 = pa0 + (size_t)16 * lda;
    const ushort_t* pb[4];
    #pragma unroll
    for (int ni = 0; ni < 4; ++ni)
        pb[ni] = BT + (size_t)(n0 + ni * 16 + l16) * ldb + 16 * hf;

    v8f acc[2][4];
    #pragma unroll
    for (int mi = 0; mi < 2; ++mi)
        #pragma unroll
        for (int ni = 0; ni < 4; ++ni) acc[mi][ni] = {};

    const int nkb = K / Kblk;
    for (int kb = 0; kb < nkb; ++kb) {
        const size_t aoff = (size_t)kb * Kblk;
        const size_t boff = (size_t)kb * kblk_stride;
        for (int i = 0; i < Kblk; i += 32) {
            __builtin_prefetch(pa0 + aoff + i + PF_ELEMS, 0, 1);
            __builtin_prefetch(pa1 + aoff + i + PF_ELEMS, 0, 1);
            #pragma unroll
            for (int ni = 0; ni < 4; ++ni)
                __builtin_prefetch(pb[ni] + boff + i + PF_ELEMS, 0, 1);

            v16bf a0 = load_a_frag(pa0 + aoff + i);
            v16bf a1 = load_a_frag(pa1 + aoff + i);
            #pragma unroll
            for (int ni = 0; ni < 4; ++ni) {
                v16bf b = load_b_frag(pb[ni] + boff + i);
                acc[0][ni] = wmma_bf16(a0, b, acc[0][ni]);
                acc[1][ni] = wmma_bf16(a1, b, acc[1][ni]);
            }
        }
    }

    #pragma unroll
    for (int mi = 0; mi < 2; ++mi) {
        #pragma unroll
        for (int ni = 0; ni < 4; ++ni) {
            const int t0 = m0 + mi * 16 + 8 * hf;
            const int n  = n0 + ni * 16 + l16;
            #pragma unroll
            for (int j = 0; j < 8; ++j) {
                const size_t idx = (size_t)(t0 + j) * ldc + n;
                float v = acc[mi][ni][j];
                if (beta) v += C[idx];
                C[idx] = v;
            }
        }
    }
}

// ---------------------------------------------------------------------------
// host side
// ---------------------------------------------------------------------------
extern "C" void kernel_launch(void* const* d_in, const int* in_sizes, int n_in,
                              void* d_out, int out_size, void* d_ws, size_t ws_size,
                              hipStream_t stream) {
    const float* x    = (const float*)d_in[0];   // [T,H]
    const float* rw   = (const float*)d_in[1];   // [H,E]
    const float* egu  = (const float*)d_in[2];   // [E,H,2I]
    const float* edw  = (const float*)d_in[3];   // [E,I,H]
    const float* sgw  = (const float*)d_in[4];   // [H,SH]
    const float* suw  = (const float*)d_in[5];   // [H,SH]
    const float* sdw  = (const float*)d_in[6];   // [SH,H]
    const float* segw = (const float*)d_in[7];   // [H,1]
    float* out = (float*)d_out;                  // [T*H] ++ [T*E]

    char* ws = (char*)d_ws;
    size_t o = 0;
    auto take = [&](size_t bytes) { size_t r = o; o += (bytes + 255) & ~(size_t)255; return r; };
    ushort_t* XB   = (ushort_t*)(ws + take((size_t)Td * Hd * 2));        // x bf16
    float*    RT   = (float*)   (ws + take((size_t)Ed * Td * 4));        // routing [E][T]
    float*    SG   = (float*)   (ws + take((size_t)Td * 4));             // sigmoid gate
    ushort_t* WG_T = (ushort_t*)(ws + take((size_t)SHd * Hd * 2));       // [SH,H]
    ushort_t* WU_T = (ushort_t*)(ws + take((size_t)SHd * Hd * 2));       // [SH,H]
    ushort_t* WD_T = (ushort_t*)(ws + take((size_t)Hd * SHd * 2));       // [H,SH]
    ushort_t* WGU_T= (ushort_t*)(ws + take((size_t)Ed * TWOI * Hd * 2)); // [E][2I,H]
    ushort_t* WDE_T= (ushort_t*)(ws + take((size_t)Ed * Hd * Id * 2));   // [E][H,I]
    ushort_t* HSH  = (ushort_t*)(ws + take((size_t)Td * SHd * 2));       // [T,SH]
    ushort_t* HEX  = (ushort_t*)(ws + take((size_t)Td * NEd * 2));       // [T,E*I]
    (void)ws_size; (void)n_in; (void)in_sizes; (void)out_size;

    // 1) x -> bf16
    {
        int n4 = (Td * Hd) / 4;
        k_cvt4<<<(n4 + 255) / 256, 256, 0, stream>>>((const float4*)x, XB, n4);
    }
    // 2) weight transposes (f32 -> bf16, transposed)
    dim3 tb(32, 8);
    k_transpose_cvt<<<dim3(SHd / 32, Hd / 32, 1),  tb, 0, stream>>>(sgw, WG_T, Hd, SHd);
    k_transpose_cvt<<<dim3(SHd / 32, Hd / 32, 1),  tb, 0, stream>>>(suw, WU_T, Hd, SHd);
    k_transpose_cvt<<<dim3(Hd / 32, SHd / 32, 1),  tb, 0, stream>>>(sdw, WD_T, SHd, Hd);
    k_transpose_cvt<<<dim3(TWOI / 32, Hd / 32, Ed), tb, 0, stream>>>(egu, WGU_T, Hd, TWOI);
    k_transpose_cvt<<<dim3(Hd / 32, Id / 32, Ed),  tb, 0, stream>>>(edw, WDE_T, Id, Hd);

    // 3) router + sigmoid shared gate; logits into tail of d_out
    k_router<<<(Td * 32) / 256, 256, 0, stream>>>(x, rw, segw, RT, SG,
                                                  out + (size_t)Td * Hd);

    // 4) shared gate/up fused (N=SH, K=H), scale = sg[t]
    k_gateup<<<dim3(SHd / 128, Td / 64), 256, 0, stream>>>(
        XB, WG_T, WU_T, SG, /*scale_blk=*/0, HSH,
        /*Nblk=*/SHd, /*blk_stride=*/0LL, /*K=*/Hd, /*ldh=*/SHd);

    // 5) expert gate/up fused (N=E*I, K=H), scale = routing[n/I][t]
    k_gateup<<<dim3(NEd / 128, Td / 64), 256, 0, stream>>>(
        XB, WGU_T, WGU_T + (size_t)Id * Hd, RT, /*scale_blk=*/Id, HEX,
        /*Nblk=*/Id, /*blk_stride=*/(long long)TWOI * Hd, /*K=*/Hd, /*ldh=*/NEd);

    // 6) shared down: out = HSH @ shared_down  (K=SH, N=H), beta=0
    k_down<<<dim3(Hd / 256, Td / 64), 256, 0, stream>>>(
        HSH, SHd, WD_T, /*ldb=*/SHd, /*Kblk=*/SHd, /*kblk_stride=*/0LL,
        /*K=*/SHd, out, Hd, /*beta=*/0);

    // 7) expert down: out += HEX @ expert_down_flat (K=E*I, N=H), beta=1
    k_down<<<dim3(Hd / 256, Td / 64), 256, 0, stream>>>(
        HEX, NEd, WDE_T, /*ldb=*/Id, /*Kblk=*/Id,
        /*kblk_stride=*/(long long)Hd * Id, /*K=*/KEd, out, Hd, /*beta=*/1);
}
